// skip_lstm_ESPnet_80513456930855
// MI455X (gfx1250) — matrix operand order repaired and checked by reading, not compiled, not run on hardware
//
#include <hip/hip_runtime.h>
#include <hip/hip_bf16.h>

typedef __bf16 bf16;
typedef __bf16 v16bf __attribute__((ext_vector_type(16)));
typedef float  v8f   __attribute__((ext_vector_type(8)));
typedef unsigned int u32x4 __attribute__((ext_vector_type(4)));

union Frag16 { v16bf v; u32x4 q[2]; };

#define B_      16
#define T_      512
#define IDIM_   320
#define CDIM_   1024
#define GDIM_   4096
#define LAYERS_ 4
#define NWG_SCAN 64

__device__ __forceinline__ float sigmoidf_(float x){ return 1.f/(1.f + __expf(-x)); }

// A fragment (16x32 bf16): lane holds 16 contiguous k from its row pointer.
__device__ __forceinline__ void load_afrag(Frag16& f, const bf16* p){
  f.q[0] = *(const u32x4*)(p);
  f.q[1] = *(const u32x4*)(p + 8);
}
// B fragment (32x16 bf16): lane holds 16 contiguous k from its row pointer.
__device__ __forceinline__ void load_bfrag(Frag16& f, const bf16* p){
  f.q[0] = *(const u32x4*)(p);
  f.q[1] = *(const u32x4*)(p + 8);
}
#define WMMA_BF16(a, b, c) __builtin_amdgcn_wmma_f32_16x16x32_bf16( \
    false, (a).v, false, (b).v, (short)0, (c), false, false)

// ---------------- elementwise helpers ----------------

__global__ void cvt_f32_bf16_k(const float* __restrict__ s, bf16* __restrict__ d, int n){
  int i = blockIdx.x*256 + threadIdx.x;
  if (i < n) d[i] = (bf16)s[i];
}

__global__ void bias_comb_k(const float* __restrict__ a, const float* __restrict__ b,
                            float* __restrict__ o, int n){
  int i = blockIdx.x*256 + threadIdx.x;
  if (i < n) o[i] = a[i] + b[i];
}

__global__ void init_scan_k(bf16* __restrict__ h, int* __restrict__ bar){
  int i = blockIdx.x*256 + threadIdx.x;
  if (i < 2*B_*CDIM_) h[i] = (bf16)0.f;
  if (i == 0) *bar = 0;
}

__global__ void copy_ilens_k(const int* __restrict__ il, int* __restrict__ dst){
  if (threadIdx.x < B_) dst[threadIdx.x] = il[threadIdx.x];
}

// ---------------- WMMA GEMM: out[m][n] = act(sum_k A[m][k]*W[n][k] + bias[n]) ----
// MODE 0: xg precompute, bf16 output, +bias
// MODE 1: projection, f32 output, mask rows by ilens, tanh(acc + bias)
// Block: 256 threads = 8 waves laid out 2(M)x4(N); wave tile = 16(M) x 64(N).
// Two named register slots rotate: loads of one slot fly while the other's
// four independent WMMAs issue. K is a template constant (multiple of 64).

template<int K, int MODE>
__global__ __launch_bounds__(256) void gemm_wmma_k(
    const bf16* __restrict__ A,
    const bf16* __restrict__ W,
    const float* __restrict__ bias,
    void* __restrict__ out, int ldo,
    const int* __restrict__ ilens)
{
  const int wave = threadIdx.x >> 5;
  const int lane = threadIdx.x & 31;
  const int lm = lane & 15, hi = lane >> 4;
  const int mbase = blockIdx.x*32 + (wave>>2)*16;
  const int nbase = blockIdx.y*256 + (wave&3)*64;

  v8f acc0 = {}, acc1 = {}, acc2 = {}, acc3 = {};

  const bf16* arow = A + (size_t)(mbase + lm)*K + hi*16;
  const bf16* w0 = W + (size_t)(nbase +  0 + lm)*K + hi*16;
  const bf16* w1 = W + (size_t)(nbase + 16 + lm)*K + hi*16;
  const bf16* w2 = W + (size_t)(nbase + 32 + lm)*K + hi*16;
  const bf16* w3 = W + (size_t)(nbase + 48 + lm)*K + hi*16;

  Frag16 faA, fbA0, fbA1, fbA2, fbA3;   // slot A
  Frag16 faB, fbB0, fbB1, fbB2, fbB3;   // slot B

  // prologue: slot A <- k=0
  load_afrag(faA, arow);
  load_bfrag(fbA0, w0); load_bfrag(fbA1, w1);
  load_bfrag(fbA2, w2); load_bfrag(fbA3, w3);

  int k0 = 0;
  for (; k0 < K - 64; k0 += 64){
    // slot B <- k0+32 (in flight during slot A's WMMAs)
    load_afrag(faB, arow + k0 + 32);
    load_bfrag(fbB0, w0 + k0 + 32); load_bfrag(fbB1, w1 + k0 + 32);
    load_bfrag(fbB2, w2 + k0 + 32); load_bfrag(fbB3, w3 + k0 + 32);
    acc0 = WMMA_BF16(faA, fbA0, acc0);
    acc1 = WMMA_BF16(faA, fbA1, acc1);
    acc2 = WMMA_BF16(faA, fbA2, acc2);
    acc3 = WMMA_BF16(faA, fbA3, acc3);
    // slot A <- k0+64 (in flight during slot B's WMMAs)
    load_afrag(faA, arow + k0 + 64);
    load_bfrag(fbA0, w0 + k0 + 64); load_bfrag(fbA1, w1 + k0 + 64);
    load_bfrag(fbA2, w2 + k0 + 64); load_bfrag(fbA3, w3 + k0 + 64);
    if (k0 + 128 < K){
      __builtin_prefetch(arow + k0 + 128, 0, 1);   // gfx1250 global_prefetch_b8
      __builtin_prefetch(w0 + k0 + 128, 0, 1);
      __builtin_prefetch(w2 + k0 + 128, 0, 1);
    }
    acc0 = WMMA_BF16(faB, fbB0, acc0);
    acc1 = WMMA_BF16(faB, fbB1, acc1);
    acc2 = WMMA_BF16(faB, fbB2, acc2);
    acc3 = WMMA_BF16(faB, fbB3, acc3);
  }
  // epilogue: last 64 K (slot A holds k0; slot B <- k0+32)
  load_afrag(faB, arow + k0 + 32);
  load_bfrag(fbB0, w0 + k0 + 32); load_bfrag(fbB1, w1 + k0 + 32);
  load_bfrag(fbB2, w2 + k0 + 32); load_bfrag(fbB3, w3 + k0 + 32);
  acc0 = WMMA_BF16(faA, fbA0, acc0);
  acc1 = WMMA_BF16(faA, fbA1, acc1);
  acc2 = WMMA_BF16(faA, fbA2, acc2);
  acc3 = WMMA_BF16(faA, fbA3, acc3);
  acc0 = WMMA_BF16(faB, fbB0, acc0);
  acc1 = WMMA_BF16(faB, fbB1, acc1);
  acc2 = WMMA_BF16(faB, fbB2, acc2);
  acc3 = WMMA_BF16(faB, fbB3, acc3);

  v8f accs[4] = {acc0, acc1, acc2, acc3};
#pragma unroll
  for (int i = 0; i < 4; ++i){
    int n = nbase + i*16 + lm;
    float bv = bias ? bias[n] : 0.f;
#pragma unroll
    for (int v = 0; v < 8; ++v){
      int m = mbase + v + 8*hi;                         // C/D layout: row = v + 8*hi
      float val = accs[i][v];
      if (MODE == 0){
        ((bf16*)out)[(size_t)m*ldo + n] = (bf16)(val + bv);
      } else {
        int b = m >> 9, t = m & (T_-1);
        if (t >= ilens[b]) val = 0.f;                   // ypad row zeroed -> tanh(bias)
        ((float*)out)[(size_t)m*ldo + n] = tanhf(val + bv);
      }
    }
  }
}

// ---------------- persistent LSTM scan: LDS-resident W_hh slab, grid-sync/step ---
// 64 WGs x 256 threads. WG owns channels [j0, j0+16). LDS holds 4 gates x 32
// k-tiles of B-fragments (128 KB) pre-swizzled to the WMMA B per-lane layout.
// Named two-slot rotation + two interleaved accumulators (no WMMA RAW chain).

__global__ __launch_bounds__(256) void lstm_scan_k(
    const bf16* __restrict__ Whh,   // [4C][C] bf16 row-major (layer slice)
    const bf16* __restrict__ xg,    // [B*T][4C] bf16 precomputed input gates
    bf16* __restrict__ ys,          // [B*T][C] layer output
    bf16* __restrict__ hbuf,        // [2][B][C] double-buffered h
    int* __restrict__ bar)
{
  extern __shared__ char smem[];
  float* red = (float*)(smem + 131072);   // 4*256 f32 partial (K upper half)
  float* pre = red + 1024;                // 4*256 f32 gate preacts

  const int wave = threadIdx.x >> 5;
  const int lane = threadIdx.x & 31;
  const int lm = lane & 15, hi = lane >> 4;
  const int j0 = blockIdx.x * 16;

  // Stage W_hh slab -> LDS in B-fragment per-lane layout (tile = 1 KB, conflict-free).
  for (int tl = wave; tl < 128; tl += 8){
    int g = tl >> 5, kt = tl & 31;
    const bf16* wr = Whh + (size_t)(g*CDIM_ + j0 + lm)*CDIM_ + kt*32 + hi*16;
    u32x4 q0 = *(const u32x4*)(wr);
    u32x4 q1 = *(const u32x4*)(wr + 8);
    u32x4* dstp = (u32x4*)(smem + tl*1024 + lane*32);
    dstp[0] = q0; dstp[1] = q1;
  }
  __syncthreads();

  const int g = wave & 3;         // gate handled by this wave
  const int khalf = wave >> 2;    // K half (0: 0..511, 1: 512..1023)
  const int em = threadIdx.x >> 4;   // elementwise: batch row
  const int en = threadIdx.x & 15;   // elementwise: channel within slice
  float c_state = 0.f;

  for (int t = 0; t < T_; ++t){
    const bf16* hin  = hbuf + (size_t)(t & 1)      * (B_*CDIM_);
    bf16*       hout = hbuf + (size_t)((t + 1) & 1)* (B_*CDIM_);

    // xg gather does not depend on h: issue first so it hides under the WMMAs.
    float xgv[8];
    if (khalf == 0){
#pragma unroll
      for (int v = 0; v < 8; ++v){
        int m = v + 8*hi;
        xgv[v] = (float)xg[(size_t)(m*T_ + t)*GDIM_ + g*CDIM_ + j0 + lm];
      }
    }

    v8f acc0 = {}, acc1 = {};
    const bf16* hrow = hin + (size_t)lm * CDIM_;
    const int ktbase = khalf * 16;

    Frag16 fa0, fb0, fa1, fb1;
    load_afrag(fa0, hrow + ktbase*32 + hi*16);
    fb0.q[0] = ((const u32x4*)(smem + (g*32 + ktbase)*1024 + lane*32))[0];
    fb0.q[1] = ((const u32x4*)(smem + (g*32 + ktbase)*1024 + lane*32))[1];

#pragma unroll
    for (int kk = 0; kk < 16; kk += 2){
      {
        const int kt = ktbase + kk + 1;
        load_afrag(fa1, hrow + kt*32 + hi*16);
        const u32x4* wp = (const u32x4*)(smem + (g*32 + kt)*1024 + lane*32);
        fb1.q[0] = wp[0]; fb1.q[1] = wp[1];
      }
      acc0 = WMMA_BF16(fa0, fb0, acc0);
      if (kk + 2 < 16){
        const int kt = ktbase + kk + 2;
        load_afrag(fa0, hrow + kt*32 + hi*16);
        const u32x4* wp = (const u32x4*)(smem + (g*32 + kt)*1024 + lane*32);
        fb0.q[0] = wp[0]; fb0.q[1] = wp[1];
      }
      acc1 = WMMA_BF16(fa1, fb1, acc1);
    }

    if (khalf == 1){
#pragma unroll
      for (int v = 0; v < 8; ++v) red[g*256 + v*32 + lane] = acc0[v] + acc1[v];
    }
    __syncthreads();
    if (khalf == 0){
#pragma unroll
      for (int v = 0; v < 8; ++v){
        float x = acc0[v] + acc1[v] + red[g*256 + v*32 + lane] + xgv[v];
        pre[g*256 + v*32 + lane] = x;
      }
    }
    __syncthreads();

    // elementwise LSTM cell update; thread owns (batch em, channel en)
    {
      int base = (em & 7)*32 + (em >> 3)*16 + en;   // C/D flat index for (em,en)
      float iv = sigmoidf_(pre[0*256 + base]);
      float fv = sigmoidf_(pre[1*256 + base]);
      float gv = tanhf    (pre[2*256 + base]);
      float ov = sigmoidf_(pre[3*256 + base]);
      c_state = fv*c_state + iv*gv;
      float hv = ov * tanhf(c_state);
      bf16 hb = (bf16)hv;
      hout[(size_t)em*CDIM_ + j0 + en] = hb;
      ys[(size_t)(em*T_ + t)*CDIM_ + j0 + en] = hb;
    }

    // grid-wide step barrier (monotonic counter; zeroed before launch)
    __threadfence();
    __syncthreads();
    if (threadIdx.x == 0){
      atomicAdd(bar, 1);
      int target = NWG_SCAN * (t + 1);
      while (*(volatile int*)bar < target) { }
    }
    __syncthreads();
    __threadfence();
  }
}

// ---------------- host orchestration ----------------

extern "C" void kernel_launch(void* const* d_in, const int* in_sizes, int n_in,
                              void* d_out, int out_size, void* d_ws, size_t ws_size,
                              hipStream_t stream)
{
  const float* xpad  = (const float*)d_in[0];
  const float* Wih0  = (const float*)d_in[1];
  const float* Wihr  = (const float*)d_in[2];
  const float* Whh   = (const float*)d_in[3];
  const float* bih   = (const float*)d_in[4];
  const float* bhh   = (const float*)d_in[5];
  const float* Wlast = (const float*)d_in[6];
  const float* blast = (const float*)d_in[7];
  const int*   ilens = (const int*)d_in[8];
  float* out = (float*)d_out;

  char* ws = (char*)d_ws;
  size_t off = 0;
  bf16* whh_b   = (bf16*)(ws+off); off += (size_t)LAYERS_*GDIM_*CDIM_*2;    // 32 MB
  bf16* wih0_b  = (bf16*)(ws+off); off += (size_t)GDIM_*IDIM_*2;            // 2.5 MB
  bf16* wihr_b  = (bf16*)(ws+off); off += (size_t)(LAYERS_-1)*GDIM_*CDIM_*2;// 24 MB
  bf16* wlast_b = (bf16*)(ws+off); off += (size_t)CDIM_*CDIM_*2;            // 2 MB
  float* biasc  = (float*)(ws+off); off += (size_t)LAYERS_*GDIM_*4;         // 64 KB
  bf16* act     = (bf16*)(ws+off); off += (size_t)B_*T_*CDIM_*2;            // 16 MB
  bf16* xg      = (bf16*)(ws+off); off += (size_t)B_*T_*GDIM_*2;            // 64 MB
  bf16* hbuf    = (bf16*)(ws+off); off += (size_t)2*B_*CDIM_*2;             // 64 KB
  int*  bar     = (int*)(ws+off);  off += 256;

  // bf16 packing of weights and layer-0 input
  cvt_f32_bf16_k<<<dim3((LAYERS_*GDIM_*CDIM_+255)/256), 256, 0, stream>>>(Whh, whh_b, LAYERS_*GDIM_*CDIM_);
  cvt_f32_bf16_k<<<dim3((GDIM_*IDIM_+255)/256), 256, 0, stream>>>(Wih0, wih0_b, GDIM_*IDIM_);
  cvt_f32_bf16_k<<<dim3(((LAYERS_-1)*GDIM_*CDIM_+255)/256), 256, 0, stream>>>(Wihr, wihr_b, (LAYERS_-1)*GDIM_*CDIM_);
  cvt_f32_bf16_k<<<dim3((CDIM_*CDIM_+255)/256), 256, 0, stream>>>(Wlast, wlast_b, CDIM_*CDIM_);
  cvt_f32_bf16_k<<<dim3((B_*T_*IDIM_+255)/256), 256, 0, stream>>>(xpad, act, B_*T_*IDIM_);
  bias_comb_k<<<dim3((LAYERS_*GDIM_+255)/256), 256, 0, stream>>>(bih, bhh, biasc, LAYERS_*GDIM_);

  for (int l = 0; l < LAYERS_; ++l){
    // xg = act @ Wih^T + (b_ih + b_hh)   [8192 x 4096]
    if (l == 0){
      gemm_wmma_k<IDIM_, 0><<<dim3(B_*T_/32, GDIM_/256), 256, 0, stream>>>(
          act, wih0_b, biasc + 0*GDIM_, xg, GDIM_, nullptr);
    } else {
      gemm_wmma_k<CDIM_, 0><<<dim3(B_*T_/32, GDIM_/256), 256, 0, stream>>>(
          act, wihr_b + (size_t)(l-1)*GDIM_*CDIM_, biasc + l*GDIM_, xg, GDIM_, nullptr);
    }
    // recurrence: persistent grid-synced scan, W_hh slab resident in LDS
    init_scan_k<<<dim3(128), 256, 0, stream>>>(hbuf, bar);
    lstm_scan_k<<<dim3(NWG_SCAN), 256, 131072 + 2*1024*4, stream>>>(
        whh_b + (size_t)l*GDIM_*CDIM_, xg, act, hbuf, bar);
  }

  // projected = tanh(mask(ys) @ W_last^T + b_last); masked rows -> tanh(b_last)
  gemm_wmma_k<CDIM_, 1><<<dim3(B_*T_/32, CDIM_/256), 256, 0, stream>>>(
      act, wlast_b, blast, out, CDIM_, ilens);

  copy_ilens_k<<<dim3(1), 32, 0, stream>>>(ilens, (int*)d_out + (size_t)B_*T_*CDIM_);
}
